// Yolo_V1_69853348102206
// MI455X (gfx1250) — matrix operand order repaired and checked by reading, not compiled, not run on hardware
//
#include <hip/hip_runtime.h>
#include <hip/hip_bf16.h>
#include <math.h>

typedef __attribute__((ext_vector_type(16))) _Float16 v16h;
typedef __attribute__((ext_vector_type(8)))  float    v8f;

#define N_BOX      4096
#define N_CLASSES  80
#define ROW_STRIDE 85
#define CONF_TH    0.01f
#define NMS_TH     0.5f
#define WORDS_PER_ROW 128   // 4096 bits / 32

// ---------------------------------------------------------------------------
// Kernel 1: decode + score. One wave (32 lanes) per prediction row.
// ---------------------------------------------------------------------------
__global__ __launch_bounds__(128) void k_decode(
    const float* __restrict__ pred,
    float* __restrict__ x1, float* __restrict__ y1,
    float* __restrict__ x2, float* __restrict__ y2,
    float* __restrict__ area, float* __restrict__ score,
    float* __restrict__ key, int* __restrict__ label, int* __restrict__ valid)
{
    const int row  = blockIdx.x * 4 + (threadIdx.x >> 5);
    const int lane = threadIdx.x & 31;
    if (row >= N_BOX) return;
    const float* p = pred + (size_t)row * ROW_STRIDE;

    // argmax + max over class logits (stable first-index tie-break)
    float v = -INFINITY; int bi = 0;
    for (int c = lane; c < N_CLASSES; c += 32) {
        float cv = p[1 + c];
        if (cv > v) { v = cv; bi = c; }    // ascending c => strict > keeps first
    }
    for (int off = 16; off; off >>= 1) {
        float ov = __shfl_xor(v, off, 32);
        int  obi = __shfl_xor(bi, off, 32);
        if (ov > v || (ov == v && obi < bi)) { v = ov; bi = obi; }
    }
    // sum of exp(c - m)  (max softmax prob = 1/sum)
    float s = 0.f;
    for (int c = lane; c < N_CLASSES; c += 32) s += expf(p[1 + c] - v);
    for (int off = 16; off; off >>= 1) s += __shfl_xor(s, off, 32);

    if (lane == 0) {
        float conf = p[0];
        float sig  = 1.f / (1.f + expf(-conf));
        float sc   = sig / s;                       // sigmoid(conf)*max(softmax)

        float b0 = p[1 + N_CLASSES + 0], b1 = p[1 + N_CLASSES + 1];
        float b2 = p[1 + N_CLASSES + 2], b3 = p[1 + N_CLASSES + 3];
        float gx = (float)(row >> 6);               // i // 64
        float gy = (float)(row & 63);               // i %  64
        float c0 = 1.f / (1.f + expf(-b0)) + gx;
        float c1 = 1.f / (1.f + expf(-b1)) + gy;
        float w  = expf(b2), h = expf(b3);
        float fx1 = fminf(fmaxf((c0 * 32.f - w) * (1.f / 2048.f), 0.f), 1.f);
        float fy1 = fminf(fmaxf((c1 * 32.f - h) * (1.f / 2048.f), 0.f), 1.f);
        float fx2 = fminf(fmaxf((c0 * 32.f + w) * (1.f / 2048.f), 0.f), 1.f);
        float fy2 = fminf(fmaxf((c1 * 32.f + h) * (1.f / 2048.f), 0.f), 1.f);

        x1[row] = fx1; y1[row] = fy1; x2[row] = fx2; y2[row] = fy2;
        area[row]  = (fy2 - fy1) * (fx2 - fx1);
        score[row] = sc;
        label[row] = bi;
        int vld    = sc > CONF_TH;
        valid[row] = vld;
        key[row]   = vld ? sc : -INFINITY;
    }
}

// ---------------------------------------------------------------------------
// Kernel 2: exact stable descending rank sort (O(N^2) counting).
// order[rank] = i, identical to jnp.argsort(-key, stable).
// ---------------------------------------------------------------------------
__global__ __launch_bounds__(256) void k_rank(
    const float* __restrict__ key, int* __restrict__ order)
{
    int i = blockIdx.x * blockDim.x + threadIdx.x;
    if (i >= N_BOX) return;
    float ki = key[i];
    int rank = 0;
    for (int j = 0; j < N_BOX; ++j) {
        float kj = key[j];
        rank += (kj > ki) || (kj == ki && j < i);
    }
    order[rank] = i;
}

// ---------------------------------------------------------------------------
// Kernel 3: fused suppression-matrix builder.
// same_class via v_wmma_f32_16x16x32_f16 on in-register one-hot fragments,
// IoU on the VALU, packed to bits via wave32 ballot -> LDS ds_or -> bitmask.
// One block (8 waves) owns a 16-row i-tile; waves stripe the 256 j-tiles.
// ---------------------------------------------------------------------------
__global__ __launch_bounds__(256) void k_supmat(
    const float* __restrict__ x1, const float* __restrict__ y1,
    const float* __restrict__ x2, const float* __restrict__ y2,
    const float* __restrict__ area, const int* __restrict__ label,
    unsigned* __restrict__ bitmask)
{
    __shared__ unsigned lw[16 * WORDS_PER_ROW];     // 16 rows x 128 words
    __shared__ float bx1[16], by1[16], bx2[16], by2[16], bar[16];

    const int ib   = blockIdx.x * 16;
    const int tid  = threadIdx.x;
    const int lane = tid & 31;
    const int wave = tid >> 5;
    const int colM = lane & 15;
    const int half = lane >> 4;

    for (int t = tid; t < 16 * WORDS_PER_ROW; t += 256) lw[t] = 0u;
    if (tid < 16) {
        bx1[tid] = x1[ib + tid]; by1[tid] = y1[ib + tid];
        bx2[tid] = x2[ib + tid]; by2[tid] = y2[ib + tid];
        bar[tid] = area[ib + tid];
    }
    __syncthreads();

    // A fragments (loop-invariant): one-hot rows of the i-tile, K padded to 96.
    // 16-bit A 16x32 layout: lanes 0-15 hold K {0..7,16..23}, lanes 16-31 hold
    // K {8..15,24..31}; VGPR v packs 2 consecutive K.
    const int albl = label[ib + colM];
    v16h afrag[3];
    for (int kc = 0; kc < 3; ++kc) {
        int kb = kc * 32;
        for (int v = 0; v < 8; ++v) {
            int k0 = kb + (v >> 2) * 16 + half * 8 + (v & 3) * 2;
            afrag[kc][2 * v]     = (albl == k0)     ? (_Float16)1.0f : (_Float16)0.0f;
            afrag[kc][2 * v + 1] = (albl == k0 + 1) ? (_Float16)1.0f : (_Float16)0.0f;
        }
    }

    for (int jt = wave; jt < N_BOX / 16; jt += 8) {
        const int jb = jt * 16;
        const int j  = jb + colM;                   // this lane's column index
        const int blbl = label[j];

        // D = A * B, B[k][n] = onehot(label[jb+n] == k).
        // 16-bit B 32x16 layout: lane = N (both halves), lanes 0-15 hold
        // K {0..7,16..23}, lanes 16-31 hold K {8..15,24..31}; VGPR v packs 2 consecutive K.
        v8f acc = {};
        for (int kc = 0; kc < 3; ++kc) {
            int kb = kc * 32;
            v16h bfrag;
            for (int v = 0; v < 8; ++v) {
                int k = kb + (v >> 2) * 16 + half * 8 + (v & 3) * 2;
                bfrag[2 * v]     = (blbl == k)     ? (_Float16)1.0f : (_Float16)0.0f;
                bfrag[2 * v + 1] = (blbl == k + 1) ? (_Float16)1.0f : (_Float16)0.0f;
            }
            acc = __builtin_amdgcn_wmma_f32_16x16x32_f16(
                false, afrag[kc], false, bfrag, (short)0, acc, false, false);
        }

        // IoU on VALU for the same 16x16 tile (j data constant over r).
        const float jx1 = x1[j], jy1 = y1[j], jx2 = x2[j], jy2 = y2[j], ja = area[j];
        const unsigned wsh = (unsigned)(jb & 16);
        const int word = jb >> 5;
        for (int r = 0; r < 8; ++r) {
            const int M = r + 8 * half;             // C/D layout: lane-half -> M+8
            const int i = ib + M;
            float xx1 = fmaxf(bx1[M], jx1);
            float yy1 = fmaxf(by1[M], jy1);
            float xx2 = fminf(bx2[M], jx2);
            float yy2 = fminf(by2[M], jy2);
            float inter = fmaxf(xx2 - xx1, 0.f) * fmaxf(yy2 - yy1, 0.f);
            float iou   = inter / (bar[M] + ja - inter);   // NaN -> false below
            bool bit = (acc[r] > 0.5f) && (iou > NMS_TH) && (i != j);
            unsigned m = __builtin_amdgcn_ballot_w32(bit);
            if (lane == 0) {
                // bits 0-15 -> row M=r, bits 16-31 -> row M=r+8, bitpos = N
                atomicOr(&lw[r * WORDS_PER_ROW + word],       (m & 0xFFFFu) << wsh);
                atomicOr(&lw[(r + 8) * WORDS_PER_ROW + word], (m >> 16)     << wsh);
            }
        }
    }
    __syncthreads();
    for (int t = tid; t < 16 * WORDS_PER_ROW; t += 256)
        bitmask[(size_t)ib * WORDS_PER_ROW + t] = lw[t];
}

// ---------------------------------------------------------------------------
// Kernel 4: serial greedy NMS scan. Single 128-thread block; the live
// suppression mask (4096 bits) stays in LDS; each accepted box ORs one
// coalesced 128-word bitmask row.
// ---------------------------------------------------------------------------
__global__ __launch_bounds__(128) void k_greedy(
    const unsigned* __restrict__ bitmask, const int* __restrict__ order,
    const int* __restrict__ valid, int* __restrict__ keep)
{
    __shared__ unsigned sup[WORDS_PER_ROW];
    __shared__ int sIdx, sActive;
    const int tid = threadIdx.x;
    sup[tid] = 0u;
    __syncthreads();

    for (int t = 0; t < N_BOX; ++t) {
        if (tid == 0) {
            int idx = order[t];
            int act = valid[idx] && !((sup[idx >> 5] >> (idx & 31)) & 1u);
            keep[idx] = act;
            sIdx = idx; sActive = act;
        }
        __syncthreads();
        if (sActive) sup[tid] |= bitmask[(size_t)sIdx * WORDS_PER_ROW + tid];
        __syncthreads();
    }
}

// ---------------------------------------------------------------------------
// Kernel 5: masked output assembly.
// d_out = [ boxes|score (4096x5) , labels (4096) , keep (4096) ] as fp32.
// ---------------------------------------------------------------------------
__global__ __launch_bounds__(256) void k_finalize(
    const float* __restrict__ x1, const float* __restrict__ y1,
    const float* __restrict__ x2, const float* __restrict__ y2,
    const float* __restrict__ score, const int* __restrict__ label,
    const int* __restrict__ keep, float* __restrict__ out)
{
    int i = blockIdx.x * blockDim.x + threadIdx.x;
    if (i >= N_BOX) return;
    float k = keep[i] ? 1.f : 0.f;
    out[i * 5 + 0] = x1[i] * k;
    out[i * 5 + 1] = y1[i] * k;
    out[i * 5 + 2] = x2[i] * k;
    out[i * 5 + 3] = y2[i] * k;
    out[i * 5 + 4] = score[i] * k;
    out[N_BOX * 5 + i] = (float)label[i];
    out[N_BOX * 6 + i] = k;
}

// ---------------------------------------------------------------------------
extern "C" void kernel_launch(void* const* d_in, const int* in_sizes, int n_in,
                              void* d_out, int out_size, void* d_ws, size_t ws_size,
                              hipStream_t stream)
{
    (void)in_sizes; (void)n_in; (void)out_size; (void)ws_size;
    const float* pred = (const float*)d_in[0];
    float* out = (float*)d_out;

    // Workspace carve-up (~2.3 MB total).
    char* ws = (char*)d_ws;
    float* x1    = (float*)ws;                 ws += N_BOX * sizeof(float);
    float* y1    = (float*)ws;                 ws += N_BOX * sizeof(float);
    float* x2    = (float*)ws;                 ws += N_BOX * sizeof(float);
    float* y2    = (float*)ws;                 ws += N_BOX * sizeof(float);
    float* area  = (float*)ws;                 ws += N_BOX * sizeof(float);
    float* score = (float*)ws;                 ws += N_BOX * sizeof(float);
    float* key   = (float*)ws;                 ws += N_BOX * sizeof(float);
    int*   label = (int*)ws;                   ws += N_BOX * sizeof(int);
    int*   valid = (int*)ws;                   ws += N_BOX * sizeof(int);
    int*   order = (int*)ws;                   ws += N_BOX * sizeof(int);
    int*   keep  = (int*)ws;                   ws += N_BOX * sizeof(int);
    unsigned* bitmask = (unsigned*)ws;         // N_BOX * 128 words (2 MB)

    k_decode  <<<N_BOX / 4, 128, 0, stream>>>(pred, x1, y1, x2, y2, area,
                                              score, key, label, valid);
    k_rank    <<<N_BOX / 256, 256, 0, stream>>>(key, order);
    k_supmat  <<<N_BOX / 16, 256, 0, stream>>>(x1, y1, x2, y2, area, label,
                                               bitmask);
    k_greedy  <<<1, WORDS_PER_ROW, 0, stream>>>(bitmask, order, valid, keep);
    k_finalize<<<N_BOX / 256, 256, 0, stream>>>(x1, y1, x2, y2, score, label,
                                                keep, out);
}
